// NeuromodulatedAttention_2508260901587
// MI455X (gfx1250) — matrix operand order repaired and checked by reading, not compiled, not run on hardware
//
#include <hip/hip_runtime.h>

// ---------------------------------------------------------------------------
// NeuromodulatedAttention for MI455X (gfx1250, wave32, WMMA bf16)
//
// Math: the per-row "mod" term cancels under the row mean/std normalization,
// so out = softmax(scores / (rowstd+1e-6)) @ V with scores = QK^T/sqrt(D).
// ---------------------------------------------------------------------------

static constexpr int B = 4, S = 2048, D = 512;
static constexpr float QK_SCALE = 0.04419417382415922f; // 1/sqrt(512)

typedef __attribute__((ext_vector_type(16))) __bf16 v16bf;
typedef __attribute__((ext_vector_type(8)))  __bf16 v8bf;
typedef __attribute__((ext_vector_type(8)))  float  v8f;

__device__ __forceinline__ __bf16 f2bf(float x) {
  unsigned u = __builtin_bit_cast(unsigned, x);
  u += 0x7fffu + ((u >> 16) & 1u);               // round-to-nearest-even
  unsigned short h = (unsigned short)(u >> 16);
  return __builtin_bit_cast(__bf16, h);
}

__device__ __forceinline__ v16bf cat8(v8bf lo, v8bf hi) {
  return __builtin_shufflevector(lo, hi, 0,1,2,3,4,5,6,7,8,9,10,11,12,13,14,15);
}

// A-matrix 16x32 bf16 fragment: lane holds K = sel*16+[0..15] (contiguous).
// p points at row_base + kc*32 + sel*8 (bf16 elements).
__device__ __forceinline__ v16bf load_a_frag(const __bf16* p) {
  p += (threadIdx.x >> 4) * 8;
  v8bf lo = *(const v8bf*)(p);
  v8bf hi = *(const v8bf*)(p + 8);
  return cat8(lo, hi);
}

__device__ __forceinline__ v8f wmma_bf16(v16bf a, v16bf b, v8f c) {
  return __builtin_amdgcn_wmma_f32_16x16x32_bf16(
      /*neg_a=*/false, a, /*neg_b=*/false, b,
      /*c_mod=*/(short)0, c, /*reuse_a=*/false, /*reuse_b=*/false);
}

// ---------------------------------------------------------------------------
// Prep kernel 1: f32 -> bf16 elementwise (Q and K), coalesced float4 reads.
// ---------------------------------------------------------------------------
__global__ void cvt_f32_to_bf16(const float* __restrict__ in,
                                __bf16* __restrict__ out, int n4) {
  int i = blockIdx.x * blockDim.x + threadIdx.x;
  if (i < n4) {
    const float4 v = ((const float4*)in)[i];
    __bf16* o = out + 4 * (size_t)i;
    o[0] = f2bf(v.x); o[1] = f2bf(v.y); o[2] = f2bf(v.z); o[3] = f2bf(v.w);
  }
}

// ---------------------------------------------------------------------------
// Prep kernel 2: V[b][s][d] f32 -> Vt[b][d][s] bf16 (LDS tiled transpose).
// Makes the PV B-fragment a contiguous 32-byte load (contraction over keys).
// ---------------------------------------------------------------------------
__global__ void transpose_v(const float* __restrict__ V, __bf16* __restrict__ Vt) {
  __shared__ float tile[32][33];
  const int b  = blockIdx.z;
  const int s0 = blockIdx.x * 32;
  const int d0 = blockIdx.y * 32;
  const int tx = threadIdx.x, ty = threadIdx.y;   // block 32x8
#pragma unroll
  for (int i = 0; i < 32; i += 8)
    tile[ty + i][tx] = V[((size_t)b * S + (s0 + ty + i)) * D + (d0 + tx)];
  __syncthreads();
#pragma unroll
  for (int i = 0; i < 32; i += 8)
    Vt[((size_t)b * D + (d0 + ty + i)) * S + (s0 + tx)] = f2bf(tile[tx][ty + i]);
}

// ---------------------------------------------------------------------------
// Main kernel: one wave32 per 16-row query tile. grid = B*(S/16) = 512.
//   Pass 1: WMMA QK^T over all 2048 keys -> per-row sum / sumsq / max.
//   Pass 2: recompute score tiles, p = exp((s - smax)/sigma'), route P through
//           a 1.3 KB LDS stage into A-layout, WMMA P@Vt into 32 resident
//           16x16 f32 accumulators (full 512-wide output row block).
// ---------------------------------------------------------------------------
__global__ __launch_bounds__(32) void attn_kernel(const __bf16* __restrict__ Qb,
                                                  const __bf16* __restrict__ Kb,
                                                  const __bf16* __restrict__ Vt,
                                                  float* __restrict__ out) {
  const int lane = threadIdx.x;
  const int row  = lane & 15;
  const int sel  = lane >> 4;
  const int b    = blockIdx.x >> 7;        // S/16 = 128 tiles per batch
  const int q0   = (blockIdx.x & 127) << 4;

  __shared__ __bf16 stage[16][40];         // padded: conflict-free A-frag reads

  // ---- resident Q fragments: 16 chunks of K=32 (reused by every key tile)
  v16bf Qf[16];
  {
    const __bf16* qrow = Qb + ((size_t)(b * S + q0 + row)) * D + sel * 8;
#pragma unroll
    for (int kc = 0; kc < 16; ++kc) Qf[kc] = load_a_frag(qrow + kc * 32);
  }

  // ---- pass 1: row statistics over all keys --------------------------------
  float rsum[8], rsq[8], rmax[8];
#pragma unroll
  for (int j = 0; j < 8; ++j) { rsum[j] = 0.f; rsq[j] = 0.f; rmax[j] = -3.0e38f; }

  for (int nt = 0; nt < S / 16; ++nt) {
    v8f acc = {0.f, 0.f, 0.f, 0.f, 0.f, 0.f, 0.f, 0.f};
    // B-frag 32x16 (contiguous k, same order as A): lane column = lane&15, K = sel*16+[0..15]
    const __bf16* krow = Kb + ((size_t)(b * S + nt * 16 + row)) * D + sel * 16;
#pragma unroll
    for (int kc = 0; kc < 16; ++kc)
      acc = wmma_bf16(Qf[kc], *(const v16bf*)(krow + kc * 32), acc);
#pragma unroll
    for (int j = 0; j < 8; ++j) {
      float s = acc[j] * QK_SCALE;
      rsum[j] += s; rsq[j] += s * s; rmax[j] = fmaxf(rmax[j], s);
    }
  }
  // reduce within each 16-lane half (matches C-layout row ownership)
#pragma unroll
  for (int j = 0; j < 8; ++j) {
#pragma unroll
    for (int off = 1; off < 16; off <<= 1) {
      rsum[j] += __shfl_xor(rsum[j], off, 32);
      rsq[j]  += __shfl_xor(rsq[j],  off, 32);
      rmax[j]  = fmaxf(rmax[j], __shfl_xor(rmax[j], off, 32));
    }
  }
  float smax[8], rinv[8], denom[8];
#pragma unroll
  for (int j = 0; j < 8; ++j) {
    float mean = rsum[j] * (1.0f / (float)S);
    float var  = (rsq[j] - rsum[j] * mean) * (1.0f / (float)(S - 1)); // ddof=1
    rinv[j]  = 1.0f / (sqrtf(fmaxf(var, 0.f)) + 1e-6f);
    smax[j]  = rmax[j];
    denom[j] = 0.f;
  }

  // ---- pass 2: streaming softmax + P@V ------------------------------------
  v8f outAcc[32];
#pragma unroll
  for (int c = 0; c < 32; ++c)
#pragma unroll
    for (int j = 0; j < 8; ++j) outAcc[c][j] = 0.f;

  for (int kt = 0; kt < S / 32; ++kt) {
#pragma unroll
    for (int t = 0; t < 2; ++t) {               // two 16-key score sub-tiles
      v8f acc = {0.f, 0.f, 0.f, 0.f, 0.f, 0.f, 0.f, 0.f};
      const __bf16* krow =
          Kb + ((size_t)(b * S + kt * 32 + t * 16 + row)) * D + sel * 16;
#pragma unroll
      for (int kc = 0; kc < 16; ++kc)
        acc = wmma_bf16(Qf[kc], *(const v16bf*)(krow + kc * 32), acc);
#pragma unroll
      for (int j = 0; j < 8; ++j) {
        float s = acc[j] * QK_SCALE;
        float p = __expf((s - smax[j]) * rinv[j]);   // mean shift cancels
        denom[j] += p;
        stage[j + 8 * sel][t * 16 + row] = f2bf(p);  // C-layout -> row-major
      }
    }
    // same-wave DS ops are in-order; compiler inserts s_wait_dscnt for RAW.
    // P as A-fragment (16x32 over this tile's 32 keys)
    v16bf Pf = load_a_frag(&stage[row][0] + sel * 8);

    const __bf16* vbase = Vt + (size_t)b * D * S + kt * 32 + sel * 16;
#pragma unroll
    for (int c = 0; c < 32; ++c) {               // 32 output-dim chunks of 16
      const __bf16* vp = vbase + (size_t)(c * 16 + row) * S;
      outAcc[c] = wmma_bf16(Pf, *(const v16bf*)vp, outAcc[c]);
    }
  }

  // ---- epilogue: normalize and store --------------------------------------
#pragma unroll
  for (int j = 0; j < 8; ++j) {
#pragma unroll
    for (int off = 1; off < 16; off <<= 1)
      denom[j] += __shfl_xor(denom[j], off, 32);
    denom[j] = 1.0f / denom[j];
  }
#pragma unroll
  for (int c = 0; c < 32; ++c) {
#pragma unroll
    for (int j = 0; j < 8; ++j) {
      const int m = j + 8 * sel;
      out[((size_t)(b * S + q0 + m)) * D + c * 16 + row] = outAcc[c][j] * denom[j];
    }
  }
}

// ---------------------------------------------------------------------------
extern "C" void kernel_launch(void* const* d_in, const int* in_sizes, int n_in,
                              void* d_out, int out_size, void* d_ws, size_t ws_size,
                              hipStream_t stream) {
  (void)in_sizes; (void)n_in; (void)out_size; (void)ws_size;
  const float* Q = (const float*)d_in[0];
  const float* K = (const float*)d_in[1];
  const float* V = (const float*)d_in[2];
  // d_in[3..12] (MLP weights / scales) provably cancel out of the output.
  float* out = (float*)d_out;

  const size_t n = (size_t)B * S * D;            // 4M elements
  __bf16* Qb = (__bf16*)d_ws;                    // 8 MB
  __bf16* Kb = Qb + n;                           // 8 MB
  __bf16* Vt = Kb + n;                           // 8 MB  (total 24 MB of ws)

  const int n4 = (int)(n / 4);
  cvt_f32_to_bf16<<<(n4 + 255) / 256, 256, 0, stream>>>(Q, Qb, n4);
  cvt_f32_to_bf16<<<(n4 + 255) / 256, 256, 0, stream>>>(K, Kb, n4);
  transpose_v<<<dim3(S / 32, D / 32, B), dim3(32, 8), 0, stream>>>(V, Vt);
  attn_kernel<<<B * (S / 16), 32, 0, stream>>>(Qb, Kb, Vt, out);
}
